// MaskedAttention_44409961840849
// MI455X (gfx1250) — matrix operand order repaired and checked by reading, not compiled, not run on hardware
//
#include <hip/hip_runtime.h>
#include <stdint.h>

// ---------------- problem constants ----------------
#define BB 4
#define SS 1024
#define WIDTH 1024
#define HH 16
#define DHH 64
#define NROWS (BB * SS)          // 4096
#define NEGV -1000000000.0f

typedef __bf16 v16bf __attribute__((ext_vector_type(16)));
typedef float  v8f   __attribute__((ext_vector_type(8)));

union ABu { v16bf v; uint32_t u[8]; };
union CDu { v8f v; float f[8]; };

__device__ __forceinline__ unsigned short f2bf(float x) {
    union { float f; uint32_t u; } v; v.f = x;
    uint32_t r = v.u + 0x7FFFu + ((v.u >> 16) & 1u);   // round-to-nearest-even
    return (unsigned short)(r >> 16);
}

// LDS byte offset of a generic pointer to a __shared__ object (for async-to-LDS vdst)
__device__ __forceinline__ uint32_t lds_offset(const void* p) {
    return (uint32_t)(uintptr_t)(__attribute__((address_space(3))) const void*)p;
}
// Async copy 8 bytes global -> LDS (tracked by ASYNCcnt), CDNA5 ISA
__device__ __forceinline__ void async_load_b64(uint32_t lds_off, const void* gsrc) {
    asm volatile("global_load_async_to_lds_b64 %0, %1, off"
                 :: "v"(lds_off), "v"((uint64_t)(uintptr_t)gsrc) : "memory");
}
__device__ __forceinline__ void wait_asynccnt0() {
    asm volatile("s_wait_asynccnt 0x0" ::: "memory");
}

// ---- WMMA bf16 fragment loaders (wave32; contiguous k order, same for A and B) ----
// A 16x32 (MxK): lane L -> m=L%16, half=L/16 ; VGPR v holds K = 16*half + 2v, +1
__device__ __forceinline__ v16bf load_a_frag(const unsigned short* __restrict__ base, int ld) {
    int lane = threadIdx.x & 31;
    int m = lane & 15, half = lane >> 4;
    const unsigned short* r = base + (size_t)m * ld;
    ABu f;
#pragma unroll
    for (int v = 0; v < 8; ++v) {
        int k = 16 * half + 2 * v;
        f.u[v] = *(const uint32_t*)(r + k);
    }
    return f.v;
}
// B 32x16 (KxN): lane L -> n=L%16, half=L/16 ; VGPR v holds K = 16*half + 2v, +1.
// Memory holds B column-major per n: element B[k][n] = base[n*ld + k] (contraction contiguous).
__device__ __forceinline__ v16bf load_b_frag(const unsigned short* __restrict__ base, int ld) {
    int lane = threadIdx.x & 31;
    int n = lane & 15, half = lane >> 4;
    const unsigned short* r = base + (size_t)n * ld + 16 * half;
    ABu f;
#pragma unroll
    for (int v = 0; v < 8; ++v) f.u[v] = *(const uint32_t*)(r + 2 * v);
    return f.v;
}

__device__ __forceinline__ v8f wmma_bf16(v16bf a, v16bf b, v8f c) {
    return __builtin_amdgcn_wmma_f32_16x16x32_bf16(false, a, false, b, (short)0, c, false, false);
}

// ---------------- fp32 -> bf16 convert ----------------
__global__ void cvt_bf16_kernel(const float* __restrict__ src, unsigned short* __restrict__ dst, int n) {
    int i = blockIdx.x * blockDim.x + threadIdx.x;
    if (i < n) dst[i] = f2bf(src[i]);
}

// ---------------- TN GEMM: Y[m,n] = sum_k A[m,k]*W[n,k] + bias[n] ----------------
// M=4096, N=1024, K=1024. Block = 8 waves (2x4), wave tile 32x64, block tile 64x256.
// mode 0: store bf16 Y[m*1024+n]; mode 1: store f32; mode 2: store bf16 transposed Vt[(b*1024+n)*1024+s]
__global__ __launch_bounds__(256) void gemm_bf16_tn(const unsigned short* __restrict__ A,
                                                    const unsigned short* __restrict__ Wt,
                                                    const float* __restrict__ bias,
                                                    void* __restrict__ out, int mode) {
    int w = threadIdx.x >> 5, lane = threadIdx.x & 31;
    int wm = w >> 2, wn = w & 3;
    int mbase = blockIdx.y * 64 + wm * 32;
    int nbase = blockIdx.x * 256 + wn * 64;

    v8f acc[2][4] = {};
    for (int k0 = 0; k0 < WIDTH; k0 += 32) {
        if (k0 + 32 < WIDTH) {  // prefetch next K tile (global_prefetch_b8)
            __builtin_prefetch(A + (size_t)mbase * WIDTH + k0 + 32, 0, 1);
            __builtin_prefetch(Wt + (size_t)nbase * WIDTH + k0 + 32, 0, 1);
        }
        v16bf a[2], b[4];
#pragma unroll
        for (int i = 0; i < 2; ++i)
            a[i] = load_a_frag(A + (size_t)(mbase + 16 * i) * WIDTH + k0, WIDTH);
#pragma unroll
        for (int j = 0; j < 4; ++j)
            b[j] = load_b_frag(Wt + (size_t)(nbase + 16 * j) * WIDTH + k0, WIDTH);
#pragma unroll
        for (int i = 0; i < 2; ++i)
#pragma unroll
            for (int j = 0; j < 4; ++j)
                acc[i][j] = wmma_bf16(a[i], b[j], acc[i][j]);
    }

    int n_l = lane & 15, rh = (lane >> 4) * 8;
#pragma unroll
    for (int i = 0; i < 2; ++i)
#pragma unroll
        for (int j = 0; j < 4; ++j) {
            int n_g = nbase + 16 * j + n_l;
            float bv = bias[n_g];
            CDu c; c.v = acc[i][j];
#pragma unroll
            for (int r = 0; r < 8; ++r) {
                int m_g = mbase + 16 * i + r + rh;
                float val = c.f[r] + bv;
                if (mode == 1) {
                    ((float*)out)[(size_t)m_g * WIDTH + n_g] = val;
                } else if (mode == 0) {
                    ((unsigned short*)out)[(size_t)m_g * WIDTH + n_g] = f2bf(val);
                } else {  // transposed: Vt[b][n][s]
                    int bb = m_g >> 10, s = m_g & 1023;
                    ((unsigned short*)out)[((size_t)(bb * WIDTH + n_g)) * SS + s] = f2bf(val);
                }
            }
        }
}

// ---------------- fused attention with head-axis softmax ----------------
// One block per (b, 16-query tile). 8 waves; wave w owns heads {2w, 2w+1}.
// Softmax over heads is local per (q,k): cross-wave via LDS.
// Mask tile is fetched with GLOBAL_LOAD_ASYNC_TO_LDS_B64, overlapped with the score WMMAs.
__global__ __launch_bounds__(256) void attn_kernel(const unsigned short* __restrict__ Qb,
                                                   const unsigned short* __restrict__ Kb,
                                                   const unsigned short* __restrict__ Vt,
                                                   const int* __restrict__ mask,
                                                   unsigned short* __restrict__ Xb) {
    __shared__ __align__(16) float          s_scores[16][32][HH];   // [q][k][h] 32 KB
    __shared__ __align__(16) unsigned short s_attn[HH][16][32];     // [h][q][k] 16 KB
    __shared__ __align__(16) int            s_mask[16][32];         // [q][k]     2 KB

    int w = threadIdx.x >> 5, lane = threadIdx.x & 31;
    int b = blockIdx.x >> 6;
    int q0 = (blockIdx.x & 63) * 16;
    const float scale = 0.125f;  // 1/sqrt(64)

    v8f acc[2][4] = {};
    v16bf aQ[2][2];
#pragma unroll
    for (int hs = 0; hs < 2; ++hs) {
        int h = 2 * w + hs;
#pragma unroll
        for (int c = 0; c < 2; ++c)
            aQ[hs][c] = load_a_frag(Qb + (size_t)(b * SS + q0) * WIDTH + h * DHH + 32 * c, WIDTH);
    }

    // per-thread mask-copy coordinates: 256 threads x 8B = 2KB tile [16 q][32 k]
    int mq = threadIdx.x >> 4;          // 0..15
    int mc = (threadIdx.x & 15) * 2;    // int pair index 0..30
    uint32_t mask_lds = lds_offset(&s_mask[mq][mc]);
    const int* mask_row = mask + (size_t)b * SS * SS + (size_t)(q0 + mq) * SS + mc;

    int n_l = lane & 15, rh = (lane >> 4) * 8;

    for (int k0 = 0; k0 < SS; k0 += 32) {
        // --- kick off async mask-tile fetch (ASYNCcnt); lands in LDS while WMMAs run ---
        async_load_b64(mask_lds, mask_row + k0);

        // --- scores S_h[16q x 32k] = Q_h (16x64) x K_h^T ---
#pragma unroll
        for (int hs = 0; hs < 2; ++hs) {
            int h = 2 * w + hs;
#pragma unroll
            for (int kh = 0; kh < 2; ++kh) {
                const unsigned short* kbase = Kb + (size_t)(b * SS + k0 + 16 * kh) * WIDTH + h * DHH;
                v16bf b0 = load_b_frag(kbase, WIDTH);
                v16bf b1 = load_b_frag(kbase + 32, WIDTH);
                v8f s = {};
                s = wmma_bf16(aQ[hs][0], b0, s);
                s = wmma_bf16(aQ[hs][1], b1, s);
                CDu cc; cc.v = s;
                int kl = 16 * kh + n_l;
#pragma unroll
                for (int r = 0; r < 8; ++r)
                    s_scores[r + rh][kl][h] = cc.f[r] * scale;
            }
        }
        wait_asynccnt0();   // mask tile resident in LDS
        __syncthreads();

        // --- softmax over the HEAD axis at each (q,k); mask broadcast over h ---
        for (int p = threadIdx.x; p < 512; p += 256) {
            int q = p >> 5, kl = p & 31;
            int mv = s_mask[q][kl];
            if (mv == 0) {
                // all scores == NEG -> uniform 1/16 over heads (faithful to reference)
                unsigned short u = f2bf(0.0625f);
#pragma unroll
                for (int h = 0; h < HH; ++h) s_attn[h][q][kl] = u;
            } else {
                float sc[HH];
#pragma unroll
                for (int h = 0; h < HH; ++h) sc[h] = s_scores[q][kl][h];
                float mx = sc[0];
#pragma unroll
                for (int h = 1; h < HH; ++h) mx = fmaxf(mx, sc[h]);
                float sum = 0.f;
#pragma unroll
                for (int h = 0; h < HH; ++h) { sc[h] = __expf(sc[h] - mx); sum += sc[h]; }
                float inv = 1.f / sum;
#pragma unroll
                for (int h = 0; h < HH; ++h) s_attn[h][q][kl] = f2bf(sc[h] * inv);
            }
        }
        __syncthreads();

        // --- out_h[16q x 64d] += attn_h (16x32) x V_h (32x64), V pre-transposed Vt[b][c][s] ---
#pragma unroll
        for (int hs = 0; hs < 2; ++hs) {
            int h = 2 * w + hs;
            v16bf aA = load_a_frag(&s_attn[h][0][0], 32);
#pragma unroll
            for (int j = 0; j < 4; ++j) {
                v16bf bV = load_b_frag(Vt + (size_t)(b * WIDTH + h * DHH + 16 * j) * SS + k0, SS);
                acc[hs][j] = wmma_bf16(aA, bV, acc[hs][j]);
            }
        }
        // next iteration's syncthreads (before s_attn rewrite) orders against these reads
    }

    // --- store X[b, q, h*64 + d] as bf16 for the output projection ---
#pragma unroll
    for (int hs = 0; hs < 2; ++hs) {
        int h = 2 * w + hs;
#pragma unroll
        for (int j = 0; j < 4; ++j) {
            CDu c; c.v = acc[hs][j];
#pragma unroll
            for (int r = 0; r < 8; ++r) {
                int q = r + rh;
                Xb[(size_t)(b * SS + q0 + q) * WIDTH + h * DHH + 16 * j + n_l] = f2bf(c.f[r]);
            }
        }
    }
}

// ---------------- host launch ----------------
extern "C" void kernel_launch(void* const* d_in, const int* in_sizes, int n_in,
                              void* d_out, int out_size, void* d_ws, size_t ws_size,
                              hipStream_t stream) {
    const float* inp  = (const float*)d_in[0];
    const int*   mask = (const int*)d_in[1];
    const float* wq = (const float*)d_in[2]; const float* bq = (const float*)d_in[3];
    const float* wk = (const float*)d_in[4]; const float* bk = (const float*)d_in[5];
    const float* wv = (const float*)d_in[6]; const float* bv = (const float*)d_in[7];
    const float* wo = (const float*)d_in[8]; const float* bo = (const float*)d_in[9];

    const size_t MBy = 1ull << 20;
    char* ws = (char*)d_ws;
    unsigned short* inpb = (unsigned short*)(ws + 0 * MBy);   // 8 MB
    unsigned short* wqb  = (unsigned short*)(ws + 8 * MBy);   // 2 MB
    unsigned short* wkb  = (unsigned short*)(ws + 10 * MBy);
    unsigned short* wvb  = (unsigned short*)(ws + 12 * MBy);
    unsigned short* wob  = (unsigned short*)(ws + 14 * MBy);
    unsigned short* Qb   = (unsigned short*)(ws + 16 * MBy);  // 8 MB
    unsigned short* Kb   = (unsigned short*)(ws + 24 * MBy);  // 8 MB
    unsigned short* Vt   = (unsigned short*)(ws + 32 * MBy);  // 8 MB (transposed [b][c][s])
    unsigned short* Xb   = (unsigned short*)(ws + 40 * MBy);  // 8 MB

    // 1) fp32 -> bf16
    int nInp = NROWS * WIDTH, nW = WIDTH * WIDTH;
    cvt_bf16_kernel<<<(nInp + 255) / 256, 256, 0, stream>>>(inp, inpb, nInp);
    cvt_bf16_kernel<<<(nW + 255) / 256, 256, 0, stream>>>(wq, wqb, nW);
    cvt_bf16_kernel<<<(nW + 255) / 256, 256, 0, stream>>>(wk, wkb, nW);
    cvt_bf16_kernel<<<(nW + 255) / 256, 256, 0, stream>>>(wv, wvb, nW);
    cvt_bf16_kernel<<<(nW + 255) / 256, 256, 0, stream>>>(wo, wob, nW);

    // 2) Q/K/V projections (V stored pre-transposed for contiguous WMMA B-frags)
    dim3 ggrid(WIDTH / 256, NROWS / 64);
    gemm_bf16_tn<<<ggrid, 256, 0, stream>>>(inpb, wqb, bq, Qb, 0);
    gemm_bf16_tn<<<ggrid, 256, 0, stream>>>(inpb, wkb, bk, Kb, 0);
    gemm_bf16_tn<<<ggrid, 256, 0, stream>>>(inpb, wvb, bv, Vt, 2);

    // 3) fused QK^T -> head-axis softmax (masked) -> attn*V
    attn_kernel<<<BB * (SS / 16), 256, 0, stream>>>(Qb, Kb, Vt, mask, Xb);

    // 4) output projection, f32 result
    gemm_bf16_tn<<<ggrid, 256, 0, stream>>>(Xb, wob, bo, d_out, 1);
}
